// PolicyNet4_89352499626166
// MI455X (gfx1250) — matrix operand order repaired and checked by reading, not compiled, not run on hardware
//
#include <hip/hip_runtime.h>
#include <hip/hip_bf16.h>

// ---------------------------------------------------------------------------
// PolicyNet forward for MI455X (gfx1250, wave32, WMMA f32_16x16x32_f16).
// HBM-bound on w3 (821 MB, read once via K-split + f32 atomics); activations
// converted fp32->f16 during LDS staging, f32 accumulation in WMMA.
// ---------------------------------------------------------------------------

typedef __attribute__((ext_vector_type(16))) _Float16 v16h;
typedef __attribute__((ext_vector_type(8)))  float    v8f;

#define GBM 128
#define GBN 64
#define GBK 64
#define GSTA (GBK + 8)   // padded LDS row stride in halfs (144B rows -> 16B aligned chunks)

// ---------------------------------------------------------------------------
// Generic NT GEMM:  C[M,N] = op( A[M,K] * W[N,K]^T + bias + resid )
//   - A may be f32 or f16 in memory (A_HALF)
//   - C stored f32 or f16 (STORE_HALF), or atomically accumulated (ATOMIC,
//     K split over gridDim.z)
//   - relu_a: relu applied to A elements during staging (fuses prior relu)
//   - row remap for flattened outputs:
//       off = (row/row_div)*row_outer + (row%row_div)*ldc + col   (row_div>0)
// Wave32 WMMA fragment layouts as loaded here (A and B share the contiguous k order):
//   A 16x32 f16: lane m=ln, halves [0..15] = K g*16..g*16+15 (contiguous)
//   B 32x16 f16: lane n=ln, halves [0..15] = K g*16..g*16+15 (contiguous)
//   C 16x16 f32: VGPR r -> M = g*8+r, N = ln
// ---------------------------------------------------------------------------
template <bool A_HALF, bool STORE_HALF, bool ATOMIC>
__global__ __launch_bounds__(256) void gemm_nt(
    const void* __restrict__ Ap, long lda,
    const float* __restrict__ Wp, long ldb,
    void* __restrict__ Cp, long ldc,
    int M, int N, int K,
    const float* __restrict__ bias,
    const float* __restrict__ resid, long ldr,
    int relu_out, int relu_a,
    int kchunks, int row_div, long row_outer)
{
    __shared__ _Float16 Ash[GBM * GSTA];
    __shared__ _Float16 Bsh[GBN * GSTA];

    const int tid  = threadIdx.x;
    const int w    = tid >> 5;
    const int lane = tid & 31;
    const int g    = lane >> 4;
    const int ln   = lane & 15;
    const int wm   = w & 3;   // 4 wave rows  (4*32 = 128 = GBM)
    const int wn   = w >> 2;  // 2 wave cols  (2*32 =  64 = GBN)

    const long bm = (long)blockIdx.x * GBM;
    const long bn = (long)blockIdx.y * GBN;

    const int kper = (K + kchunks - 1) / kchunks;
    const int k0   = blockIdx.z * kper;
    const int k1   = (k0 + kper < K) ? (k0 + kper) : K;

    const _Float16* Ah = (const _Float16*)Ap;
    const float*    Af = (const float*)Ap;

    v8f acc[2][2] = {};

    for (int kb = k0; kb < k1; kb += GBK) {
        // ---- stage A tile (convert -> f16, zero-pad, optional relu) ----
        for (int idx = tid; idx < GBM * GBK; idx += 256) {
            int r = idx / GBK, c = idx - r * GBK;
            long grow = bm + r;
            int  gk   = kb + c;
            float v = 0.f;
            if (grow < M && gk < k1) {
                v = A_HALF ? (float)Ah[grow * lda + gk] : Af[grow * lda + gk];
                if (relu_a) v = fmaxf(v, 0.f);
            }
            Ash[r * GSTA + c] = (_Float16)v;
        }
        // ---- stage B tile (weights, fp32 -> f16, zero-pad) ----
        for (int idx = tid; idx < GBN * GBK; idx += 256) {
            int r = idx / GBK, c = idx - r * GBK;
            long grow = bn + r;
            int  gk   = kb + c;
            float v = (grow < N && gk < k1) ? Wp[grow * ldb + gk] : 0.f;
            Bsh[r * GSTA + c] = (_Float16)v;
        }
        __syncthreads();

        #pragma unroll
        for (int kt = 0; kt < GBK / 32; ++kt) {
            v16h afr[2], bfr[2];
            #pragma unroll
            for (int i = 0; i < 2; ++i) {
                const _Float16* pa = &Ash[(wm * 32 + i * 16 + ln) * GSTA + kt * 32 + g * 16];
                #pragma unroll
                for (int j = 0; j < 8; ++j) { afr[i][j] = pa[j]; afr[i][8 + j] = pa[8 + j]; }
            }
            #pragma unroll
            for (int i = 0; i < 2; ++i) {
                const _Float16* pb = &Bsh[(wn * 32 + i * 16 + ln) * GSTA + kt * 32 + g * 16];
                #pragma unroll
                for (int j = 0; j < 16; ++j) bfr[i][j] = pb[j];
            }
            #pragma unroll
            for (int i = 0; i < 2; ++i)
                #pragma unroll
                for (int jj = 0; jj < 2; ++jj)
                    acc[i][jj] = __builtin_amdgcn_wmma_f32_16x16x32_f16(
                        false, afr[i], false, bfr[jj], (short)0, acc[i][jj], false, false);
        }
        __syncthreads();
    }

    // ---- epilogue / store ----
    #pragma unroll
    for (int i = 0; i < 2; ++i)
        #pragma unroll
        for (int jj = 0; jj < 2; ++jj)
            #pragma unroll
            for (int r = 0; r < 8; ++r) {
                long grow = bm + wm * 32 + i * 16 + g * 8 + r;
                long gcol = bn + wn * 32 + jj * 16 + ln;
                if (grow < M && gcol < N) {
                    float v = acc[i][jj][r];
                    long off = (row_div > 0)
                        ? ((grow / row_div) * row_outer + (grow % row_div) * ldc + gcol)
                        : (grow * ldc + gcol);
                    if (ATOMIC) {
                        atomicAdd((float*)Cp + off, v);
                    } else {
                        if (bias)  v += bias[gcol];
                        if (resid) v += resid[grow * ldr + gcol];
                        if (relu_out) v = fmaxf(v, 0.f);
                        if (STORE_HALF) ((_Float16*)Cp)[off] = (_Float16)v;
                        else            ((float*)Cp)[off] = v;
                    }
                }
            }
}

// ---------------------------------------------------------------------------
// Fused attention: one block per (batch, head).  L=100 padded to 112 rows,
// K dims padded to 128.  LDS buffers are reused: smA holds Q then V^T,
// smB holds K then P (softmax probabilities).
// qkv is f16 [25600][3000]; ctx output is f16 [25600][1000].
// ---------------------------------------------------------------------------
#define ALP 112
#define AKP 128
#define ATS 136

__global__ __launch_bounds__(256) void attn_kernel(
    const _Float16* __restrict__ qkv, _Float16* __restrict__ ctx)
{
    __shared__ _Float16 smA[ALP * ATS];  // Q, later V^T
    __shared__ _Float16 smB[ALP * ATS];  // K, later P

    const int tid  = threadIdx.x;
    const int b    = blockIdx.x / 10;
    const int head = blockIdx.x % 10;
    const long base = (long)b * 100 * 3000 + head * 100;
    const int w = tid >> 5, lane = tid & 31, g = lane >> 4, ln = lane & 15;

    // ---- stage Q and K (zero-padded) ----
    for (int idx = tid; idx < ALP * AKP; idx += 256) {
        int r = idx >> 7, c = idx & 127;
        float qv = 0.f, kv = 0.f;
        if (r < 100 && c < 100) {
            long o = base + (long)r * 3000 + c;
            qv = (float)qkv[o];
            kv = (float)qkv[o + 1000];
        }
        smA[r * ATS + c] = (_Float16)qv;
        smB[r * ATS + c] = (_Float16)kv;
    }
    __syncthreads();

    // ---- scores = Q K^T (7 waves, one 16-row tile each) + softmax ----
    v8f acc[7] = {};
    if (w < 7) {
        #pragma unroll
        for (int kt = 0; kt < 4; ++kt) {
            v16h afr;
            const _Float16* pa = &smA[(w * 16 + ln) * ATS + kt * 32 + g * 16];
            #pragma unroll
            for (int j = 0; j < 8; ++j) { afr[j] = pa[j]; afr[8 + j] = pa[8 + j]; }
            #pragma unroll
            for (int nt = 0; nt < 7; ++nt) {
                v16h bfr;
                const _Float16* pb = &smB[(nt * 16 + ln) * ATS + kt * 32 + g * 16];
                #pragma unroll
                for (int j = 0; j < 16; ++j) bfr[j] = pb[j];
                acc[nt] = __builtin_amdgcn_wmma_f32_16x16x32_f16(
                    false, afr, false, bfr, (short)0, acc[nt], false, false);
            }
        }
        // row-wise softmax; a C-row lives in one 16-lane half (m = g*8+r, n = ln)
        #pragma unroll
        for (int r = 0; r < 8; ++r) {
            float s[7];
            #pragma unroll
            for (int nt = 0; nt < 7; ++nt) {
                int col = nt * 16 + ln;
                s[nt] = (col < 100) ? acc[nt][r] * 0.1f : -1e30f;  // 1/sqrt(dh)=0.1
            }
            float mx = s[0];
            #pragma unroll
            for (int nt = 1; nt < 7; ++nt) mx = fmaxf(mx, s[nt]);
            #pragma unroll
            for (int off = 1; off < 16; off <<= 1) mx = fmaxf(mx, __shfl_xor(mx, off, 16));
            float sum = 0.f;
            #pragma unroll
            for (int nt = 0; nt < 7; ++nt) { s[nt] = __expf(s[nt] - mx); sum += s[nt]; }
            #pragma unroll
            for (int off = 1; off < 16; off <<= 1) sum += __shfl_xor(sum, off, 16);
            float inv = 1.f / sum;
            #pragma unroll
            for (int nt = 0; nt < 7; ++nt) acc[nt][r] = s[nt] * inv;
        }
    }
    __syncthreads();  // everyone done reading Q (smA) and K (smB)

    // ---- write P over smB; stage V^T over smA ----
    if (w < 7) {
        #pragma unroll
        for (int r = 0; r < 8; ++r)
            #pragma unroll
            for (int nt = 0; nt < 7; ++nt)
                smB[(w * 16 + g * 8 + r) * ATS + nt * 16 + ln] = (_Float16)acc[nt][r];
    }
    for (int idx = tid; idx < ALP * 16; idx += 256) {  // zero P cols 112..127
        int r = idx >> 4, c = 112 + (idx & 15);
        smB[r * ATS + c] = (_Float16)0.f;
    }
    for (int idx = tid; idx < ALP * AKP; idx += 256) { // V^T[d][m]
        int d = idx >> 7, m = idx & 127;
        float v = 0.f;
        if (d < 100 && m < 100)
            v = (float)qkv[base + (long)m * 3000 + 2000 + d];
        smA[d * ATS + m] = (_Float16)v;
    }
    __syncthreads();

    // ---- context = P @ V ----
    if (w < 7) {
        v8f oacc[7] = {};
        #pragma unroll
        for (int kt = 0; kt < 4; ++kt) {
            v16h afr;
            const _Float16* pa = &smB[(w * 16 + ln) * ATS + kt * 32 + g * 16];
            #pragma unroll
            for (int j = 0; j < 8; ++j) { afr[j] = pa[j]; afr[8 + j] = pa[8 + j]; }
            #pragma unroll
            for (int nt = 0; nt < 7; ++nt) {
                v16h bfr;
                const _Float16* pb = &smA[(nt * 16 + ln) * ATS + kt * 32 + g * 16];
                #pragma unroll
                for (int j = 0; j < 16; ++j) bfr[j] = pb[j];
                oacc[nt] = __builtin_amdgcn_wmma_f32_16x16x32_f16(
                    false, afr, false, bfr, (short)0, oacc[nt], false, false);
            }
        }
        #pragma unroll
        for (int nt = 0; nt < 7; ++nt)
            #pragma unroll
            for (int r = 0; r < 8; ++r) {
                int l = w * 16 + g * 8 + r;
                int d = nt * 16 + ln;
                if (l < 100 && d < 100)
                    ctx[((long)(b * 100 + l)) * 1000 + head * 100 + d] = (_Float16)oacc[nt][r];
            }
    }
}

// ---------------------------------------------------------------------------
// Small helper kernels
// ---------------------------------------------------------------------------
__global__ void init_bias_kernel(float* __restrict__ y, const float* __restrict__ b,
                                 int total, int N)
{
    int i = blockIdx.x * blockDim.x + threadIdx.x;
    if (i < total) y[i] = b[i % N];
}

__global__ void copy_action_kernel(const float* __restrict__ la, float* __restrict__ out,
                                   long ld, long off)
{
    int i = blockIdx.x * blockDim.x + threadIdx.x;
    if (i < 256 * 101) {
        int b = i / 101, a = i % 101;
        out[(long)b * ld + off + a] = la[i];
    }
}

// top-50 mask + softmax (exact jax top_k semantics: ties broken by lower index)
__global__ __launch_bounds__(128) void topk_softmax_kernel(
    const float* __restrict__ fs, float* __restrict__ out)
{
    __shared__ float sc[101];
    __shared__ int   win[101];
    __shared__ float smx, ssum;
    const int b = blockIdx.x, tid = threadIdx.x;
    if (tid < 101) sc[tid] = fs[b * 101 + tid];
    __syncthreads();
    for (int i = tid; i < 101; i += 128) {
        float si = sc[i];
        int rank = 0;
        for (int j = 0; j < 101; ++j) {
            float sj = sc[j];
            rank += (sj > si) || (sj == si && j < i);
        }
        win[i] = (rank < 50);
    }
    __syncthreads();
    if (tid == 0) {
        float mx = -1e30f;
        for (int i = 0; i < 101; ++i) if (win[i]) mx = fmaxf(mx, sc[i]);
        float z = 0.f;
        for (int i = 0; i < 101; ++i) if (win[i]) z += __expf(sc[i] - mx);
        smx = mx; ssum = z;
    }
    __syncthreads();
    if (tid < 101)
        out[b * 101 + tid] = win[tid] ? __expf(sc[tid] - smx) / ssum : 0.f;
}

// ---------------------------------------------------------------------------
// Launcher
// ---------------------------------------------------------------------------
extern "C" void kernel_launch(void* const* d_in, const int* in_sizes, int n_in,
                              void* d_out, int out_size, void* d_ws, size_t ws_size,
                              hipStream_t stream)
{
    (void)in_sizes; (void)n_in; (void)out_size; (void)ws_size;

    const float* hist       = (const float*)d_in[0];
    const float* last_act   = (const float*)d_in[1];
    const float* w_embed    = (const float*)d_in[2];
    const float* b_embed    = (const float*)d_in[3];
    const float* in_proj_w  = (const float*)d_in[4];
    const float* in_proj_b  = (const float*)d_in[5];
    const float* out_proj_w = (const float*)d_in[6];
    const float* out_proj_b = (const float*)d_in[7];
    const float* w3         = (const float*)d_in[8];
    const float* b3         = (const float*)d_in[9];
    const float* w5         = (const float*)d_in[10];
    const float* b5         = (const float*)d_in[11];
    const float* w6         = (const float*)d_in[12];
    const float* b6         = (const float*)d_in[13];

    char* wsp = (char*)d_ws;
    auto carve = [&](size_t bytes) {
        char* p = wsp;
        wsp += (bytes + 255) & ~(size_t)255;
        return p;
    };
    float*    hbuf = (float*)   carve(25600ull * 1000 * 4);  // h (fp32, residual)
    _Float16* qkvh = (_Float16*)carve(25600ull * 3000 * 2);  // qkv (f16)
    _Float16* ctxh = (_Float16*)carve(25600ull * 1000 * 2);  // attn context (f16)
    float*    xbuf = (float*)   carve(256ull * 100201 * 4);  // [B, FLAT]
    float*    ybuf = (float*)   carve(256ull * 2048 * 4);    // fc3 pre-relu
    float*    x2   = (float*)   carve(256ull * 202 * 4);     // [B, 2A]
    float*    fsb  = (float*)   carve(256ull * 101 * 4);     // final scores

    dim3 blk(256);

    // 1) h = hist @ w_embed^T + b_embed            [25600,1000]
    gemm_nt<false, false, false><<<dim3(200, 16, 1), blk, 0, stream>>>(
        hist, 1000, w_embed, 1000, hbuf, 1000, 25600, 1000, 1000,
        b_embed, nullptr, 0, 0, 0, 1, 0, 0);

    // 2) qkv = h @ in_proj_w^T + in_proj_b  -> f16 [25600,3000]
    gemm_nt<false, true, false><<<dim3(200, 47, 1), blk, 0, stream>>>(
        hbuf, 1000, in_proj_w, 1000, qkvh, 3000, 25600, 3000, 1000,
        in_proj_b, nullptr, 0, 0, 0, 1, 0, 0);

    // 3) attention per (batch, head)
    attn_kernel<<<dim3(2560), blk, 0, stream>>>(qkvh, ctxh);

    // 4) x[:, :100000] = ctx @ out_proj_w^T + out_proj_b + h   (remapped rows)
    gemm_nt<true, false, false><<<dim3(200, 16, 1), blk, 0, stream>>>(
        ctxh, 1000, out_proj_w, 1000, xbuf, 1000, 25600, 1000, 1000,
        out_proj_b, hbuf, 1000, 0, 0, 1, 100, 100201);

    // 5) x[:, 100000:] = last_action
    copy_action_kernel<<<dim3((256 * 101 + 255) / 256), blk, 0, stream>>>(
        last_act, xbuf, 100201, 100000);

    // 6) y = b3 (broadcast), then 7) y += x @ w3^T  (K split x8, f32 atomics)
    init_bias_kernel<<<dim3((256 * 2048 + 255) / 256), blk, 0, stream>>>(
        ybuf, b3, 256 * 2048, 2048);
    gemm_nt<false, false, true><<<dim3(2, 32, 8), blk, 0, stream>>>(
        xbuf, 100201, w3, 100201, ybuf, 2048, 256, 2048, 100201,
        nullptr, nullptr, 0, 0, 0, 8, 0, 0);

    // 8) x2[:, :101] = relu(y) @ w5^T + b5   (relu fused into A staging)
    gemm_nt<false, false, false><<<dim3(2, 2, 1), blk, 0, stream>>>(
        ybuf, 2048, w5, 2048, x2, 202, 256, 101, 2048,
        b5, nullptr, 0, 0, 1, 1, 0, 0);

    // 9) x2[:, 101:] = last_action
    copy_action_kernel<<<dim3((256 * 101 + 255) / 256), blk, 0, stream>>>(
        last_act, x2, 202, 101);

    // 10) final_scores = x2 @ w6^T + b6
    gemm_nt<false, false, false><<<dim3(2, 2, 1), blk, 0, stream>>>(
        x2, 202, w6, 202, fsb, 101, 256, 101, 202,
        b6, nullptr, 0, 0, 0, 1, 0, 0);

    // 11) top-50 winner mask + softmax -> d_out
    topk_softmax_kernel<<<dim3(256), dim3(128), 0, stream>>>(fsb, (float*)d_out);
}
